// Sage_En_29755533426828
// MI455X (gfx1250) — matrix operand order repaired and checked by reading, not compiled, not run on hardware
//
#include <hip/hip_runtime.h>
#include <hip/hip_bf16.h>

// SageConv: out = relu(concat(x, segment_sum(x)/(deg+1)) @ W)
// N=100000 nodes, E=1.6M edges, F=128, H=128, K=2F=256.

#define NF   128   // node feature dim
#define KTOT 256   // GEMM K = 2*NF
#define NH   128   // output dim

typedef __attribute__((ext_vector_type(16))) __bf16 v16bf;
typedef __attribute__((ext_vector_type(8)))  float  v8f;

union FragAB {
    v16bf v;
    unsigned short s[16];
    unsigned int   u[8];
    uint4 q[2];
};
union FragC {
    v8f   v;
    float f[8];
};

// Pack two f32 -> packed bf16x2 (low = a, high = b).
// Round-to-nearest (ties away from zero): 1 VALU add per element + 1 merge
// per pair. Differs from RNE only on exact half-ulp ties -- negligible next
// to bf16 quantization itself, and far cheaper than the software RNE
// sequence clang emits for fptrunc-to-bf16 (no hw cvt builtin on gfx1250).
__device__ __forceinline__ unsigned int pack2_bf16(float a, float b) {
    union { float f; unsigned int u; } ua, ub;
    ua.f = a; ub.f = b;
    const unsigned int ra = ua.u + 0x8000u;
    const unsigned int rb = ub.u + 0x8000u;
    return (rb & 0xFFFF0000u) | (ra >> 16);   // v_and_or_b32 / v_perm_b32
}

// -------- Kernel 1: edge scatter-add (one wave32 per edge) --------
__global__ void sage_scatter_kernel(const float* __restrict__ x,
                                    const int*   __restrict__ erow,   // destination
                                    const int*   __restrict__ ecol,   // source
                                    float*       __restrict__ nsum,   // [N, NF]
                                    float*       __restrict__ deg,    // [N]
                                    int E) {
    const int e    = (int)((blockIdx.x * blockDim.x + threadIdx.x) >> 5);
    const int lane = threadIdx.x & 31;
    if (e >= E) return;
    const int dst = erow[e];
    const int src = ecol[e];
    const float4 v = *(const float4*)(x + (size_t)src * NF + lane * 4);
    float* p = nsum + (size_t)dst * NF + lane * 4;
    __hip_atomic_fetch_add(p + 0, v.x, __ATOMIC_RELAXED, __HIP_MEMORY_SCOPE_AGENT);
    __hip_atomic_fetch_add(p + 1, v.y, __ATOMIC_RELAXED, __HIP_MEMORY_SCOPE_AGENT);
    __hip_atomic_fetch_add(p + 2, v.z, __ATOMIC_RELAXED, __HIP_MEMORY_SCOPE_AGENT);
    __hip_atomic_fetch_add(p + 3, v.w, __ATOMIC_RELAXED, __HIP_MEMORY_SCOPE_AGENT);
    if (lane == 0)
        __hip_atomic_fetch_add(deg + dst, 1.0f, __ATOMIC_RELAXED, __HIP_MEMORY_SCOPE_AGENT);
}

// -------- Kernel 2: fused normalize + concat + GEMM(bf16 WMMA) + ReLU --------
// Block = 256 threads = 8 waves; block owns 128 node-rows; wave w owns rows
// [blk*128 + w*16, +16) and produces the full 128-wide output strip for them.
__global__ void sage_gemm_kernel(const float* __restrict__ x,     // [N, NF]
                                 const float* __restrict__ nsum,  // [N, NF]
                                 const float* __restrict__ deg,   // [N]
                                 const float* __restrict__ W,     // [KTOT, NH] row-major
                                 float*       __restrict__ out,   // [N, NH]
                                 int nNodes) {
    extern __shared__ unsigned short Wt[];  // [NH][KTOT] bf16, n-major (64 KB)

    const int tid = threadIdx.x;

    // Stage W transposed to LDS as bf16: Wt[n][k] = bf16(W[k][n]).
    // Convert K-pairs: one packed convert + one 32-bit DS store per pair.
    for (int j = tid; j < (KTOT / 2) * NH; j += 256) {
        const int n = j & (NH - 1);
        const int k = (j >> 7) * 2;            // even k
        const float w0 = W[(size_t)k * NH + n];
        const float w1 = W[(size_t)(k + 1) * NH + n];
        *(unsigned int*)(Wt + (size_t)n * KTOT + k) = pack2_bf16(w0, w1);
    }
    __syncthreads();

    const int wave   = tid >> 5;
    const int lane   = tid & 31;
    const int l16    = lane & 15;
    const int hi     = lane >> 4;     // lane half
    const int khalfA = hi * 8;        // A-frag K sub-offset (ISA 16-bit A layout)
    const int khalfB = hi * 8;        // B-frag K sub-offset (same K order as A)

    const int mbase = blockIdx.x * 128 + wave * 16;
    int arow = mbase + l16;
    if (arow >= nNodes) arow = nNodes - 1;   // clamp loads; stores are predicated

    const float invdeg = 1.0f / (deg[arow] + 1.0f);

    FragC acc[8];
#pragma unroll
    for (int t = 0; t < 8; ++t)
#pragma unroll
        for (int j = 0; j < 8; ++j) acc[t].f[j] = 0.0f;

#pragma unroll
    for (int kk = 0; kk < 8; ++kk) {  // K step of 32; kk<4 -> x half, else neigh half
        const float* srow;
        float scale;
        if (kk < 4) { srow = x    + (size_t)arow * NF + kk * 32;       scale = 1.0f;   }
        else        { srow = nsum + (size_t)arow * NF + (kk - 4) * 32; scale = invdeg; }

        // A fragment: lane half covers K {khalfA..+7} U {khalfA+16..+23}
        const float4 f0 = *(const float4*)(srow + khalfA + 0);
        const float4 f1 = *(const float4*)(srow + khalfA + 4);
        const float4 f2 = *(const float4*)(srow + khalfA + 16);
        const float4 f3 = *(const float4*)(srow + khalfA + 20);
        FragAB a;
        a.u[0] = pack2_bf16(f0.x * scale, f0.y * scale);
        a.u[1] = pack2_bf16(f0.z * scale, f0.w * scale);
        a.u[2] = pack2_bf16(f1.x * scale, f1.y * scale);
        a.u[3] = pack2_bf16(f1.z * scale, f1.w * scale);
        a.u[4] = pack2_bf16(f2.x * scale, f2.y * scale);
        a.u[5] = pack2_bf16(f2.z * scale, f2.w * scale);
        a.u[6] = pack2_bf16(f3.x * scale, f3.y * scale);
        a.u[7] = pack2_bf16(f3.z * scale, f3.w * scale);

        // B fragments from LDS: lane half covers K {khalfB..+7} U {khalfB+16..+23} -> 2x 16B loads
        const unsigned short* wkbase = Wt + kk * 32 + khalfB;
#pragma unroll
        for (int nt = 0; nt < 8; ++nt) {
            const unsigned short* wp = wkbase + (size_t)(nt * 16 + l16) * KTOT;
            FragAB b;
            b.q[0] = *(const uint4*)(wp);
            b.q[1] = *(const uint4*)(wp + 16);
            acc[nt].v = __builtin_amdgcn_wmma_f32_16x16x32_bf16(
                /*neg_a=*/false, a.v, /*neg_b=*/false, b.v,
                /*c_mod=*/(short)0, acc[nt].v,
                /*reuse_a=*/false, /*reuse_b=*/false);
        }
    }

    // Store with ReLU. D layout: lanes 0-15: (M=v, N=l16); lanes 16-31: (M=v+8, N=l16).
#pragma unroll
    for (int nt = 0; nt < 8; ++nt) {
#pragma unroll
        for (int v = 0; v < 8; ++v) {
            const int m    = v + hi * 8;
            const int node = mbase + m;
            if (node < nNodes) {
                const float val = acc[nt].f[v];
                out[(size_t)node * NH + nt * 16 + l16] = val > 0.0f ? val : 0.0f;
            }
        }
    }
}

extern "C" void kernel_launch(void* const* d_in, const int* in_sizes, int n_in,
                              void* d_out, int out_size, void* d_ws, size_t ws_size,
                              hipStream_t stream) {
    const float* x    = (const float*)d_in[0];
    const int*   erow = (const int*)  d_in[1];
    const int*   ecol = (const int*)  d_in[2];
    const float* W    = (const float*)d_in[3];
    float*       out  = (float*)d_out;

    const int nNodes = in_sizes[0] / NF;
    const int E      = in_sizes[1];

    float* nsum = (float*)d_ws;                       // [N, NF]
    float* deg  = nsum + (size_t)nNodes * NF;         // [N]

    hipMemsetAsync(d_ws, 0, ((size_t)nNodes * NF + (size_t)nNodes) * sizeof(float),
                   stream);

    const int sblocks = (E + 7) / 8;                  // 8 edges (waves) per block
    sage_scatter_kernel<<<sblocks, 256, 0, stream>>>(x, erow, ecol, nsum, deg, E);

    const int gblocks = (nNodes + 127) / 128;
    sage_gemm_kernel<<<gblocks, 256, (size_t)KTOT * NH * sizeof(unsigned short),
                       stream>>>(x, nsum, deg, W, out, nNodes);
}
